// Divide_Cross_Attention_Transformer_52982716563827
// MI455X (gfx1250) — matrix operand order repaired and checked by reading, not compiled, not run on hardware
//
#include <hip/hip_runtime.h>
#include <hip/hip_bf16.h>
#include <math.h>
#include <stdint.h>

#define HWSZ   65536
#define BATCH  4
#define NHEADS 4
#define EPSLN  1e-5f

typedef __attribute__((ext_vector_type(16))) _Float16 v16h;
typedef __attribute__((ext_vector_type(8)))  float    v8f;

// ---------------------------------------------------------------------------
// WMMA fragment helpers (wave32, v_wmma_f32_16x16x32_f16 layouts per CDNA5 ISA)
// A 16x32 f16 from row-major [M][K]: lane m<16 holds K{0..7,16..23} (contig),
//   lane m+16 holds K{8..15,24..31}. -> 2 x 16B contiguous LDS loads.
// B 32x16 f16 from *[col][k] row-major* storage: lane (half,n) holds
//   K{8*half+0..7, 16+8*half+0..7} of column col0+n (same order as A).
// C/D 16x16 f32: vgpr r, lane l -> row r+8*(l>>4), col l&15
// ---------------------------------------------------------------------------
__device__ __forceinline__ v16h load_a_frag(const _Float16* base, int row0, int k0, int ld) {
  const int l = threadIdx.x & 31;
  const int hlf = l >> 4;
  const int m = l & 15;
  const _Float16* p = base + (row0 + m) * ld + k0 + 8 * hlf;
  v16h a;
#pragma unroll
  for (int i = 0; i < 8; ++i) { a[i] = p[i]; a[i + 8] = p[16 + i]; }
  return a;
}

// B operand stored transposed: S[col][k], row pitch ld (halves).
__device__ __forceinline__ v16h load_b_frag_t(const _Float16* base, int k0, int col0, int ld) {
  const int l = threadIdx.x & 31;
  const int hlf = l >> 4;
  const int n = l & 15;
  const _Float16* p = base + (col0 + n) * ld + k0 + 8 * hlf;
  v16h b;
#pragma unroll
  for (int i = 0; i < 8; ++i) { b[i] = p[i]; b[i + 8] = p[16 + i]; }
  return b;
}

__device__ __forceinline__ v8f wmma16(v16h a, v16h b, v8f c) {
  return __builtin_amdgcn_wmma_f32_16x16x32_f16(false, a, false, b, (short)0, c, false, false);
}

// ---------------------------------------------------------------------------
// Async global -> LDS DMA (CDNA5, ASYNCcnt-tracked). 16B per lane per op.
// LDS destination byte offset = low 32 bits of the generic shared pointer
// (VDST holds the LDS byte address, taken relative to the LDS base, ISA 10.2).
// ---------------------------------------------------------------------------
__device__ __forceinline__ void async_b128(const void* gptr, void* lptr) {
  unsigned loff = (unsigned)(uintptr_t)lptr;
  asm volatile("global_load_async_to_lds_b128 %0, %1, off"
               :: "v"(loff), "v"((unsigned long long)(uintptr_t)gptr)
               : "memory");
}
__device__ __forceinline__ void wait_async() {
  asm volatile("s_wait_asynccnt 0" ::: "memory");
}

// ---------------------------------------------------------------------------
// K0: per-pixel channel-LayerNorm statistics for image / event / common / diff
// stats layout (floats): [8][B*HW] : mu_i, rs_i, mu_e, rs_e, mu_c, rs_c, mu_d, rs_d
// ---------------------------------------------------------------------------
__global__ __launch_bounds__(256) void stats_kernel(const float* __restrict__ img,
                                                    const float* __restrict__ evt,
                                                    float* __restrict__ stats) {
  const int pid = blockIdx.x * 256 + threadIdx.x;       // 0 .. B*HW-1
  const int b = pid / HWSZ;
  const int n = pid % HWSZ;
  const float* ip = img + (size_t)b * 64 * HWSZ + n;
  const float* ep = evt + (size_t)b * 64 * HWSZ + n;
  float si = 0.f, se = 0.f, ssi = 0.f, sse = 0.f, sie = 0.f;
#pragma unroll 8
  for (int c = 0; c < 64; ++c) {
    float a = ip[(size_t)c * HWSZ];
    float e = ep[(size_t)c * HWSZ];
    si += a; se += e; ssi += a * a; sse += e * e; sie += a * e;
  }
  const float inv = 1.f / 64.f;
  const int BHW = BATCH * HWSZ;
  float mu, var;
  mu = si * inv;          var = ssi * inv - mu * mu;
  stats[0 * BHW + pid] = mu; stats[1 * BHW + pid] = rsqrtf(var + EPSLN);
  mu = se * inv;          var = sse * inv - mu * mu;
  stats[2 * BHW + pid] = mu; stats[3 * BHW + pid] = rsqrtf(var + EPSLN);
  mu = (si + se) * inv;   var = (ssi + 2.f * sie + sse) * inv - mu * mu;
  stats[4 * BHW + pid] = mu; stats[5 * BHW + pid] = rsqrtf(var + EPSLN);
  mu = (si - se) * inv;   var = (ssi - 2.f * sie + sse) * inv - mu * mu;
  stats[6 * BHW + pid] = mu; stats[7 * BHW + pid] = rsqrtf(var + EPSLN);
}

__global__ __launch_bounds__(256) void zero_kernel(float* __restrict__ p, int n) {
  int i = blockIdx.x * 256 + threadIdx.x;
  if (i < n) p[i] = 0.f;
}

// ---------------------------------------------------------------------------
// K0b: convert all GEMM weights to f16 once (dense row-major) so the per-tile
// kernels can stage them into LDS with async b128 DMA.
// layout (halves): wqc[4096] wkc[4096] wqd[4096] wkd[4096]
//                  fc1c[8192] fc2c[8192] fc1d[8192] fc2d[8192]   (49152 total)
// ---------------------------------------------------------------------------
__global__ __launch_bounds__(256) void cvt_weights_kernel(
    const float* __restrict__ wqc, const float* __restrict__ wkc,
    const float* __restrict__ wqd, const float* __restrict__ wkd,
    const float* __restrict__ f1c, const float* __restrict__ f2c,
    const float* __restrict__ f1d, const float* __restrict__ f2d,
    _Float16* __restrict__ dst) {
  int i = blockIdx.x * 256 + threadIdx.x;   // 0 .. 49151
  float v;
  if (i < 4096)        v = wqc[i];
  else if (i < 8192)   v = wkc[i - 4096];
  else if (i < 12288)  v = wqd[i - 8192];
  else if (i < 16384)  v = wkd[i - 12288];
  else if (i < 24576)  v = f1c[i - 16384];
  else if (i < 32768)  v = f2c[i - 24576];
  else if (i < 40960)  v = f1d[i - 32768];
  else                 v = f2d[i - 40960];
  dst[i] = (_Float16)v;
}

// ---------------------------------------------------------------------------
// K1: per-tile q/k GEMMs (WMMA) + global Gram / sum-of-squares accumulation
// gram layout: [4 slots: ic,ec,id,ed][B][HEADS][16][16]
// sumsq layout: [6 slots: qc,kic,kec,qd,kid,ked][B][64]
// ---------------------------------------------------------------------------
__global__ __launch_bounds__(256) void qk_gram_kernel(
    const float* __restrict__ img, const float* __restrict__ evt,
    const _Float16* __restrict__ wf,   // f16 weights (cvt_weights layout)
    const float* __restrict__ lw0, const float* __restrict__ lb0,
    const float* __restrict__ lw1, const float* __restrict__ lb1,
    const float* __restrict__ lw2, const float* __restrict__ lb2,
    const float* __restrict__ lw3, const float* __restrict__ lb3,
    const float* __restrict__ stats, float* __restrict__ gram, float* __restrict__ sumsq) {
  __shared__ alignas(16) unsigned char smem[50432];
  float*    s_st = (float*)smem;               // [8][64] mu/rs per tensor
  float*    s_lw = s_st + 512;                 // [8][64] w/b per tensor
  _Float16* s_wq = (_Float16*)(s_lw + 512);    // [64][72] row-major (A operand)
  _Float16* s_wk = s_wq + 64 * 72;
  _Float16* s_x  = s_wk + 64 * 72;             // [pix][ch] (B operand, transposed)
  _Float16* s_q  = s_x + 64 * 72;              // [ch][pix] (A operand of Gram)
  _Float16* s_k  = s_q + 64 * 72;              // [ch][pix] (B operand of Gram, [col][k])
  float*    s_tmp = (float*)(s_k + 64 * 72);   // [64]

  const int t = threadIdx.x;
  const int b = blockIdx.y;
  const int n0 = blockIdx.x * 64;
  const int wv = t >> 5;
  const int l = t & 31, hlf = l >> 4, nn = l & 15;
  const int BHW = BATCH * HWSZ;

  for (int idx = t; idx < 512; idx += 256) {
    int arr = idx >> 6, j = idx & 63;
    s_st[idx] = stats[arr * BHW + b * HWSZ + n0 + j];
    const float* src;
    switch (arr) {
      case 0: src = lw0; break; case 1: src = lb0; break;
      case 2: src = lw1; break; case 3: src = lb1; break;
      case 4: src = lw2; break; case 5: src = lb2; break;
      case 6: src = lw3; break; default: src = lb3; break;
    }
    s_lw[idx] = src[j];
  }

  float ri[16], re[16];
#pragma unroll
  for (int it = 0; it < 16; ++it) {
    int idx = it * 256 + t;
    int c = idx >> 6, n = idx & 63;
    size_t g = (size_t)(b * 64 + c) * HWSZ + n0 + n;
    ri[it] = img[g];
    re[it] = evt[g];
  }
  __syncthreads();

  for (int br = 0; br < 2; ++br) {
    // async DMA the two weight matrices (f16, dense [64][64]) into padded LDS
    const _Float16* wq_g = wf + br * 8192;
    const _Float16* wk_g = wq_g + 4096;
    for (int idx = t; idx < 512; idx += 256) {   // 512 chunks of 16B each
      int row = idx >> 3, cc = idx & 7;
      async_b128(wq_g + idx * 8, s_wq + row * 72 + cc * 8);
      async_b128(wk_g + idx * 8, s_wk + row * 72 + cc * 8);
    }
    { // build query-source tile (nc or nd), pixel-major [pix][ch]
      int mode = br ? 3 : 2;
#pragma unroll
      for (int it = 0; it < 16; ++it) {
        int idx = it * 256 + t;
        int c = idx >> 6, n = idx & 63;
        float v = br ? (ri[it] - re[it]) : (ri[it] + re[it]);
        float o = (v - s_st[mode * 128 + n]) * s_st[mode * 128 + 64 + n] *
                      s_lw[mode * 128 + c] + s_lw[mode * 128 + 64 + c];
        s_x[n * 72 + c] = (_Float16)o;
      }
    }
    wait_async();
    __syncthreads();
    // q = Wq @ X   (64x64 @ 64x64, 16 tiles / 8 waves); store channel-major
    for (int j = 0; j < 2; ++j) {
      int tile = wv * 2 + j, mt = tile >> 2, nt = tile & 3;
      v8f acc = {};
      for (int kc = 0; kc < 64; kc += 32)
        acc = wmma16(load_a_frag(s_wq, mt * 16, kc, 72),
                     load_b_frag_t(s_x, kc, nt * 16, 72), acc);
#pragma unroll
      for (int r = 0; r < 8; ++r)
        s_q[(mt * 16 + r + 8 * hlf) * 72 + nt * 16 + nn] = (_Float16)acc[r];
    }
    __syncthreads();
    { // sumsq(q rows) -> slot br*3
      int c = t & 63, qq = t >> 6;
      float s = 0.f;
      for (int p = qq * 16; p < qq * 16 + 16; ++p) {
        float v = (float)s_q[c * 72 + p];
        s += v * v;
      }
      if (t < 64) s_tmp[t] = 0.f;
      __syncthreads();
      atomicAdd(&s_tmp[c], s);
      __syncthreads();
      if (t < 64) atomicAdd(&sumsq[(br * 3 + 0) * 256 + b * 64 + t], s_tmp[t]);
    }
    __syncthreads();

    for (int si = 0; si < 2; ++si) {
      { // build key-source tile (ni or ne), pixel-major
        int mode = si;
#pragma unroll
        for (int it = 0; it < 16; ++it) {
          int idx = it * 256 + t;
          int c = idx >> 6, n = idx & 63;
          float v = si ? re[it] : ri[it];
          float o = (v - s_st[mode * 128 + n]) * s_st[mode * 128 + 64 + n] *
                        s_lw[mode * 128 + c] + s_lw[mode * 128 + 64 + c];
          s_x[n * 72 + c] = (_Float16)o;
        }
      }
      __syncthreads();
      // k = Wk @ X, stored channel-major [ch][pix] (== [col][k] for Gram's B)
      for (int j = 0; j < 2; ++j) {
        int tile = wv * 2 + j, mt = tile >> 2, nt = tile & 3;
        v8f acc = {};
        for (int kc = 0; kc < 64; kc += 32)
          acc = wmma16(load_a_frag(s_wk, mt * 16, kc, 72),
                       load_b_frag_t(s_x, kc, nt * 16, 72), acc);
#pragma unroll
        for (int r = 0; r < 8; ++r)
          s_k[(mt * 16 + r + 8 * hlf) * 72 + nt * 16 + nn] = (_Float16)acc[r];
      }
      __syncthreads();
      { // sumsq(k rows) -> slot br*3+1+si
        int c = t & 63, qq = t >> 6;
        float s = 0.f;
        for (int p = qq * 16; p < qq * 16 + 16; ++p) {
          float v = (float)s_k[c * 72 + p];
          s += v * v;
        }
        if (t < 64) s_tmp[t] = 0.f;
        __syncthreads();
        atomicAdd(&s_tmp[c], s);
        __syncthreads();
        if (t < 64) atomicAdd(&sumsq[(br * 3 + 1 + si) * 256 + b * 64 + t], s_tmp[t]);
      }
      // Gram per head: q_h [16 x 64pix] x k_h^T -> atomic accumulate 16x16
      if (wv < 4) {
        int h = wv;
        v8f acc = {};
        for (int kc = 0; kc < 64; kc += 32)
          acc = wmma16(load_a_frag(s_q, h * 16, kc, 72),
                       load_b_frag_t(s_k, kc, h * 16, 72), acc);
        int slot = br * 2 + si;
#pragma unroll
        for (int r = 0; r < 8; ++r)
          atomicAdd(&gram[((slot * BATCH + b) * NHEADS + h) * 256 + (r + 8 * hlf) * 16 + nn],
                    acc[r]);
      }
      __syncthreads();
    }
  }
}

// ---------------------------------------------------------------------------
// K2: normalize Grams, softmax, combine, fold E = proj @ blockdiag(A) @ Wv
// One block per (batch, type), type 0 = common, 1 = diff-img, 2 = diff-evt.
// ---------------------------------------------------------------------------
__device__ __forceinline__ void softmax16(float* r) {
  float m = r[0];
  for (int d = 1; d < 16; ++d) m = fmaxf(m, r[d]);
  float s = 0.f;
  for (int d = 0; d < 16; ++d) { r[d] = expf(r[d] - m); s += r[d]; }
  float inv = 1.f / s;
  for (int d = 0; d < 16; ++d) r[d] *= inv;
}

__global__ __launch_bounds__(256) void attn_e_kernel(
    const float* __restrict__ gram, const float* __restrict__ sumsq,
    const float* __restrict__ temp_c, const float* __restrict__ temp_d,
    const float* __restrict__ Wv_c, const float* __restrict__ Wv_d,
    const float* __restrict__ proj1, const float* __restrict__ proj2,
    _Float16* __restrict__ Ews) {
  __shared__ float s_a[64][16];
  __shared__ float s_T[64][66];
  const int bt = blockIdx.x;
  const int b = bt / 3, type = bt % 3;
  const int t = threadIdx.x;

  if (t < 64) {
    int h = t >> 4, cl = t & 15;
    int g1, qs, k1;
    const float* temp;
    if (type == 0)      { g1 = 0; qs = 0; k1 = 1; temp = temp_c; }
    else if (type == 1) { g1 = 2; qs = 3; k1 = 4; temp = temp_d; }
    else                { g1 = 3; qs = 3; k1 = 5; temp = temp_d; }
    float qn = fmaxf(sqrtf(sumsq[qs * 256 + b * 64 + t]), 1e-12f);
    float tp = temp[h];
    float row[16];
    for (int d = 0; d < 16; ++d) {
      float kn = fmaxf(sqrtf(sumsq[k1 * 256 + b * 64 + h * 16 + d]), 1e-12f);
      row[d] = gram[((g1 * BATCH + b) * NHEADS + h) * 256 + cl * 16 + d] / (qn * kn) * tp;
    }
    softmax16(row);
    if (type == 0) {  // common: attn = softmax(ic) * softmax(ec)
      float row2[16];
      for (int d = 0; d < 16; ++d) {
        float kn = fmaxf(sqrtf(sumsq[2 * 256 + b * 64 + h * 16 + d]), 1e-12f);
        row2[d] = gram[((1 * BATCH + b) * NHEADS + h) * 256 + cl * 16 + d] / (qn * kn) * tp;
      }
      softmax16(row2);
      for (int d = 0; d < 16; ++d) row[d] *= row2[d];
    }
    for (int d = 0; d < 16; ++d) s_a[t][d] = row[d];
  }
  __syncthreads();
  const float* Wv = (type == 0) ? Wv_c : Wv_d;
  for (int idx = t; idx < 4096; idx += 256) {  // T = blockdiag(A) @ Wv
    int c = idx >> 6, j = idx & 63, h = c >> 4;
    float s = 0.f;
    for (int d = 0; d < 16; ++d) s += s_a[c][d] * Wv[(h * 16 + d) * 64 + j];
    s_T[c][j] = s;
  }
  __syncthreads();
  const float* P = (type == 0) ? proj1 : proj2;
  for (int idx = t; idx < 4096; idx += 256) {  // E = proj @ T
    int o = idx >> 6, j = idx & 63;
    float s = 0.f;
    for (int c = 0; c < 64; ++c) s += P[o * 64 + c] * s_T[c][j];
    Ews[((size_t)(b * 3 + type) * 64 + o) * 64 + j] = (_Float16)s;
  }
}

// ---------------------------------------------------------------------------
// K3: apply folded E matrices + per-pixel LN + FFN (all GEMMs on WMMA), store.
// ---------------------------------------------------------------------------
__global__ __launch_bounds__(256) void apply_ffn_kernel(
    const float* __restrict__ img, const float* __restrict__ evt,
    const float* __restrict__ stats,
    const float* __restrict__ lw0, const float* __restrict__ lb0,
    const float* __restrict__ lw1, const float* __restrict__ lb1,
    const float* __restrict__ ln1w, const float* __restrict__ ln1b,
    const float* __restrict__ ln2w, const float* __restrict__ ln2b,
    const float* __restrict__ fc1cb, const float* __restrict__ fc2cb,
    const float* __restrict__ fc1db, const float* __restrict__ fc2db,
    const _Float16* __restrict__ wf, const _Float16* __restrict__ Ews,
    float* __restrict__ out) {
  __shared__ alignas(16) unsigned char smem[61952];
  // phase-1 views
  float*    s_st = (float*)smem;                 // [4][64]  mu/rs img,evt
  float*    s_lw = s_st + 256;                   // [4][64]  w/b img,evt
  _Float16* s_E  = (_Float16*)(s_lw + 256);      // 3 x [64][64] (A operands)
  _Float16* s_x  = s_E + 12288;                  // [pix][ch] (B operand)
  // phase-2 views (reuse, phases separated by barriers)
  float*    s_Y  = (float*)smem;                 // [64ch][66pix] f32
  _Float16* s_YN = (_Float16*)(smem + 16896);    // [64pix][72] f16 (A operand)
  _Float16* s_WT = (_Float16*)(smem + 26112);    // B operands [col][k], <= 18432 B
  _Float16* s_H  = (_Float16*)(smem + 44544);    // [64pix][136] f16 (A operand)

  const int t = threadIdx.x;
  const int b = blockIdx.y;
  const int n0 = blockIdx.x * 64;
  const int wv = t >> 5;
  const int l = t & 31, hlf = l >> 4, nn = l & 15;
  const int BHW = BATCH * HWSZ;

  // async DMA the three per-batch E matrices (dense f16, 24576 B)
  const _Float16* EwsB = Ews + (size_t)b * 12288;
  for (int idx = t; idx < 1536; idx += 256)
    async_b128(EwsB + idx * 8, s_E + idx * 8);

  {
    int arr = t >> 6, j = t & 63;
    s_st[t] = stats[arr * BHW + b * HWSZ + n0 + j];
    const float* src = (arr == 0) ? lw0 : (arr == 1) ? lb0 : (arr == 2) ? lw1 : lb1;
    s_lw[t] = src[j];
  }

  float ri[16], re[16];
#pragma unroll
  for (int it = 0; it < 16; ++it) {
    int idx = it * 256 + t;
    int c = idx >> 6, n = idx & 63;
    size_t g = (size_t)(b * 64 + c) * HWSZ + n0 + n;
    ri[it] = img[g];
    re[it] = evt[g];
  }
  wait_async();
  __syncthreads();

  v8f ac[2] = {};
  v8f ad[2] = {};
  // -------- phase 1: out_c += E_c@ni ; out_d += E_id@ni --------
#pragma unroll
  for (int it = 0; it < 16; ++it) {
    int idx = it * 256 + t;
    int c = idx >> 6, n = idx & 63;
    float o = (ri[it] - s_st[n]) * s_st[64 + n] * s_lw[c] + s_lw[64 + c];
    s_x[n * 72 + c] = (_Float16)o;
  }
  __syncthreads();
  for (int j = 0; j < 2; ++j) {
    int tile = wv * 2 + j, mt = tile >> 2, nt = tile & 3;
    for (int kc = 0; kc < 64; kc += 32) {
      v16h bf = load_b_frag_t(s_x, kc, nt * 16, 72);
      ac[j] = wmma16(load_a_frag(s_E, mt * 16, kc, 64), bf, ac[j]);
      ad[j] = wmma16(load_a_frag(s_E + 4096, mt * 16, kc, 64), bf, ad[j]);
    }
  }
  __syncthreads();
  // -------- out_c += E_c@ne ; out_d += E_ed@ne --------
#pragma unroll
  for (int it = 0; it < 16; ++it) {
    int idx = it * 256 + t;
    int c = idx >> 6, n = idx & 63;
    float o = (re[it] - s_st[128 + n]) * s_st[192 + n] * s_lw[128 + c] + s_lw[192 + c];
    s_x[n * 72 + c] = (_Float16)o;
  }
  __syncthreads();
  for (int j = 0; j < 2; ++j) {
    int tile = wv * 2 + j, mt = tile >> 2, nt = tile & 3;
    for (int kc = 0; kc < 64; kc += 32) {
      v16h bf = load_b_frag_t(s_x, kc, nt * 16, 72);
      ac[j] = wmma16(load_a_frag(s_E, mt * 16, kc, 64), bf, ac[j]);
      ad[j] = wmma16(load_a_frag(s_E + 8192, mt * 16, kc, 64), bf, ad[j]);
    }
  }
  __syncthreads();

  // -------- phase 2: per-branch LN + FFN + residual + store --------
  for (int br = 0; br < 2; ++br) {
    const float* lnw = br ? ln2w : ln1w;
    const float* lnb = br ? ln2b : ln1b;
    const float* f1b = br ? fc1db : fc1cb;
    const float* f2b = br ? fc2db : fc2cb;
    const _Float16* f1h = wf + 16384 + br * 16384;  // fc1 [128][64] f16
    const _Float16* f2h = wf + 24576 + br * 16384;  // fc2 [64][128] f16

    for (int j = 0; j < 2; ++j) {  // dump accumulators [ch][pix]
      int tile = wv * 2 + j, mt = tile >> 2, nt = tile & 3;
      v8f acc = br ? ad[j] : ac[j];
#pragma unroll
      for (int r = 0; r < 8; ++r)
        s_Y[(mt * 16 + r + 8 * hlf) * 66 + nt * 16 + nn] = acc[r];
    }
    __syncthreads();
    // async DMA fc1 into [128][72] while LN runs
    for (int idx = t; idx < 1024; idx += 256) {
      int row = idx >> 3, cc = idx & 7;
      async_b128(f1h + idx * 8, s_WT + row * 72 + cc * 8);
    }
    if (t < 64) {  // LayerNorm over channels at pixel t, write pixel-major f16
      float s = 0.f, ss = 0.f;
      for (int c = 0; c < 64; ++c) {
        float v = s_Y[c * 66 + t];
        s += v; ss += v * v;
      }
      float mu = s * (1.f / 64.f);
      float var = ss * (1.f / 64.f) - mu * mu;
      float rs = rsqrtf(var + EPSLN);
      for (int c = 0; c < 64; ++c) {
        float v = (s_Y[c * 66 + t] - mu) * rs * lnw[c] + lnb[c];
        s_YN[t * 72 + c] = (_Float16)v;
      }
    }
    wait_async();
    __syncthreads();
    // H = gelu(YN @ fc1^T + b1) : 4x8 = 32 tiles / 8 waves
    for (int j = 0; j < 4; ++j) {
      int tile = wv * 4 + j, mt = tile >> 3, nt = tile & 7;
      v8f acc = {};
      for (int kc = 0; kc < 64; kc += 32)
        acc = wmma16(load_a_frag(s_YN, mt * 16, kc, 72),
                     load_b_frag_t(s_WT, kc, nt * 16, 72), acc);
      int col = nt * 16 + nn;
      float bias = f1b[col];
#pragma unroll
      for (int r = 0; r < 8; ++r) {
        float x = acc[r] + bias;
        float g = 0.5f * x * (1.f + erff(x * 0.70710678118f));  // exact-erf GELU
        s_H[(mt * 16 + r + 8 * hlf) * 136 + col] = (_Float16)g;
      }
    }
    __syncthreads();
    // async DMA fc2 into [64][136]
    for (int idx = t; idx < 1024; idx += 256) {
      int row = idx >> 4, cc = idx & 15;
      async_b128(f2h + idx * 8, s_WT + row * 136 + cc * 8);
    }
    wait_async();
    __syncthreads();
    // O = H @ fc2^T + b2 + residual ; store [b,c,h,w]
    size_t obase = (size_t)br * BATCH * 64 * HWSZ;
    for (int j = 0; j < 2; ++j) {
      int tile = wv * 2 + j, mt = tile >> 2, nt = tile & 3;
      v8f acc = {};
      for (int kc = 0; kc < 128; kc += 32)
        acc = wmma16(load_a_frag(s_H, mt * 16, kc, 136),
                     load_b_frag_t(s_WT, kc, nt * 16, 136), acc);
      int c = nt * 16 + nn;
      float bias = f2b[c];
#pragma unroll
      for (int r = 0; r < 8; ++r) {
        int pix = mt * 16 + r + 8 * hlf;
        float val = acc[r] + bias + s_Y[c * 66 + pix];
        out[obase + (size_t)(b * 64 + c) * HWSZ + n0 + pix] = val;
      }
    }
    __syncthreads();
  }
}

// ---------------------------------------------------------------------------
// workspace layout (floats):
//   [0, 2097152)              stats   8 x B*HW
//   [2097152, 2113536)        gram    4 x B x 4 x 256
//   [2113536, 2115072)        sumsq   6 x B x 64
//   [2115072, 2139648)        E (f16) B x 3 x 64 x 64        (49152 halves)
//   [2139648, 2164224)        weights (f16)                  (49152 halves)
// ---------------------------------------------------------------------------
extern "C" void kernel_launch(void* const* d_in, const int* in_sizes, int n_in,
                              void* d_out, int out_size, void* d_ws, size_t ws_size,
                              hipStream_t stream) {
  (void)in_sizes; (void)n_in; (void)out_size; (void)ws_size;
  const float* img   = (const float*)d_in[0];
  const float* evt   = (const float*)d_in[1];
  const float* Wq_c  = (const float*)d_in[2];
  const float* Wk_c  = (const float*)d_in[3];
  const float* Wv_c  = (const float*)d_in[4];
  const float* tmp_c = (const float*)d_in[5];
  const float* Wq_d  = (const float*)d_in[6];
  const float* Wk_d  = (const float*)d_in[7];
  const float* Wv_d  = (const float*)d_in[8];
  const float* tmp_d = (const float*)d_in[9];
  const float* lniw  = (const float*)d_in[10];
  const float* lnib  = (const float*)d_in[11];
  const float* lnew  = (const float*)d_in[12];
  const float* lneb  = (const float*)d_in[13];
  const float* lncw  = (const float*)d_in[14];
  const float* lncb  = (const float*)d_in[15];
  const float* lndw  = (const float*)d_in[16];
  const float* lndb  = (const float*)d_in[17];
  const float* ln1w  = (const float*)d_in[18];
  const float* ln1b  = (const float*)d_in[19];
  const float* ln2w  = (const float*)d_in[20];
  const float* ln2b  = (const float*)d_in[21];
  const float* fc1c  = (const float*)d_in[22];
  const float* fc1cb = (const float*)d_in[23];
  const float* fc2c  = (const float*)d_in[24];
  const float* fc2cb = (const float*)d_in[25];
  const float* fc1d  = (const float*)d_in[26];
  const float* fc1db = (const float*)d_in[27];
  const float* fc2d  = (const float*)d_in[28];
  const float* fc2db = (const float*)d_in[29];
  const float* proj1 = (const float*)d_in[30];
  const float* proj2 = (const float*)d_in[31];

  float* ws    = (float*)d_ws;
  float* stats = ws;
  float* gram  = ws + 2097152;
  float* sumsq = gram + 16384;
  _Float16* Ews = (_Float16*)(sumsq + 1536);
  _Float16* wf  = Ews + 49152;
  float* out = (float*)d_out;

  zero_kernel<<<70, 256, 0, stream>>>(gram, 16384 + 1536);
  cvt_weights_kernel<<<192, 256, 0, stream>>>(Wq_c, Wk_c, Wq_d, Wk_d,
                                              fc1c, fc2c, fc1d, fc2d, wf);
  stats_kernel<<<(BATCH * HWSZ) / 256, 256, 0, stream>>>(img, evt, stats);
  qk_gram_kernel<<<dim3(HWSZ / 64, BATCH), 256, 0, stream>>>(
      img, evt, wf,
      lniw, lnib, lnew, lneb, lncw, lncb, lndw, lndb,
      stats, gram, sumsq);
  attn_e_kernel<<<BATCH * 3, 256, 0, stream>>>(
      gram, sumsq, tmp_c, tmp_d, Wv_c, Wv_d, proj1, proj2, Ews);
  apply_ffn_kernel<<<dim3(HWSZ / 64, BATCH), 256, 0, stream>>>(
      img, evt, stats, lniw, lnib, lnew, lneb,
      ln1w, ln1b, ln2w, ln2b,
      fc1cb, fc2cb, fc1db, fc2db,
      wf, Ews, out);
}
